// ViTBlock_74406013435974
// MI455X (gfx1250) — matrix operand order repaired and checked by reading, not compiled, not run on hardware
//
#include <hip/hip_runtime.h>
#include <cstddef>

typedef __attribute__((ext_vector_type(16))) __bf16 v16bf;
typedef __attribute__((ext_vector_type(8)))  __bf16 v8bf;
typedef __attribute__((ext_vector_type(4)))  __bf16 v4bf;
typedef __attribute__((ext_vector_type(8)))  float  v8f;
typedef __attribute__((ext_vector_type(4)))  int    v4i;
typedef __attribute__((ext_vector_type(8)))  int    v8i;
typedef __attribute__((ext_vector_type(4)))  unsigned int v4u;

#define DIM      1024
#define SEQ      2048
#define BATCH    4
#define NHEADS   16
#define HDIM     64
#define MLPDIM   4096
#define TOKENS   (BATCH * SEQ)
#define LDST     40   // 32 elems + 8 pad (16B) -> matches TDM pad descriptor

union U16 { v8bf h[2]; v16bf v; };

// ---------------------------------------------------------------------------
// TDM: async 2D tile load (128 rows x 32 bf16) global -> LDS with +16B row pad.
// D# group0: count=1 | lds_addr | global_addr(57b) | type=2
// D# group1: data_size=2B, pad_enable, pad_interval=16dw, pad_amount=4dw,
//            tensor_dim0=K, tensor_dim1=rows, tile=32x128, dim0_stride=K
// ---------------------------------------------------------------------------
__device__ __forceinline__ void tdm_load_tile(unsigned lds_off,
                                              const __bf16* gaddr,
                                              int K, int rows) {
  const size_t ga = (size_t)gaddr;
  const unsigned kk = (unsigned)K, rr = (unsigned)rows;
  v4u g0;
  g0[0] = 1u;                                            // count=1
  g0[1] = lds_off;                                       // lds byte offset
  g0[2] = (unsigned)(ga & 0xffffffffu);                  // global addr lo
  g0[3] = (unsigned)((ga >> 32) & 0x1ffffffu) | 0x80000000u; // addr hi | type=2
  v8i g1;
  g1[0] = (int)((1u << 16) | (1u << 20) | (3u << 22) | (3u << 25));
  g1[1] = (int)(kk << 16);                               // tensor_dim0 lo16
  g1[2] = (int)((kk >> 16) | (rr << 16));                // dim0 hi | dim1 lo
  g1[3] = (int)((rr >> 16) | (32u << 16));               // dim1 hi | tile_dim0=32
  g1[4] = 128;                                           // tile_dim1=128
  g1[5] = (int)kk;                                       // dim0_stride lo32
  g1[6] = 0;
  g1[7] = 0;
  v4i gz4 = {0, 0, 0, 0};
  v8i gz8 = {0, 0, 0, 0, 0, 0, 0, 0};
  __builtin_amdgcn_tensor_load_to_lds(g0, g1, gz4, gz4, gz8, 0);
}

// ---------------------------------------------------------------------------
// Weight transpose + fp32 -> bf16:  W[K][N] -> Wt[N][K]
// ---------------------------------------------------------------------------
__global__ __launch_bounds__(256) void transpose_w_bf16(
    const float* __restrict__ W, __bf16* __restrict__ Wt, int K, int N) {
  __shared__ float tile[32][33];
  const int tx = threadIdx.x & 31, ty = threadIdx.x >> 5;   // 32x8
  const int n0 = blockIdx.x * 32, k0 = blockIdx.y * 32;
#pragma unroll
  for (int i = 0; i < 4; ++i)
    tile[ty + i * 8][tx] = W[(size_t)(k0 + ty + i * 8) * N + (n0 + tx)];
  __syncthreads();
#pragma unroll
  for (int i = 0; i < 4; ++i)
    Wt[(size_t)(n0 + ty + i * 8) * K + (k0 + tx)] = (__bf16)tile[tx][ty + i * 8];
}

// ---------------------------------------------------------------------------
// RMSNorm (fp32 in) -> bf16 out, one block per token row of 1024
// ---------------------------------------------------------------------------
__global__ __launch_bounds__(256) void rmsnorm_bf16(
    const float* __restrict__ x, const float* __restrict__ w,
    __bf16* __restrict__ out) {
  const int row = blockIdx.x, t = threadIdx.x;
  const float4 v = ((const float4*)(x + (size_t)row * DIM))[t];
  float ss = v.x * v.x + v.y * v.y + v.z * v.z + v.w * v.w;
#pragma unroll
  for (int m = 16; m >= 1; m >>= 1) ss += __shfl_xor(ss, m, 32);
  __shared__ float red[8];
  if ((t & 31) == 0) red[t >> 5] = ss;
  __syncthreads();
  float tot = 0.f;
#pragma unroll
  for (int i = 0; i < 8; ++i) tot += red[i];
  const float scale = rsqrtf(tot * (1.0f / DIM) + 1e-6f);
  const float4 ww = ((const float4*)w)[t];
  v4bf o;
  o[0] = (__bf16)(v.x * scale * ww.x);
  o[1] = (__bf16)(v.y * scale * ww.y);
  o[2] = (__bf16)(v.z * scale * ww.z);
  o[3] = (__bf16)(v.w * scale * ww.w);
  *(v4bf*)(out + (size_t)row * DIM + t * 4) = o;
}

// ---------------------------------------------------------------------------
// Tiled bf16 WMMA GEMM, TDM double-buffered tiles.
// C[M,N] = A[M,K] * Bt[N,K]^T, fp32 accumulate.
// EPI 0: store bf16 raw | 1: fp32 (acc+bias+resid) | 2: bf16 silu(acc+bias)
// ---------------------------------------------------------------------------
template <int EPI>
__global__ __launch_bounds__(256) void gemm_bf16_wmma(
    const __bf16* __restrict__ A, const __bf16* __restrict__ Bt,
    const float* __restrict__ bias, const float* __restrict__ resid,
    float* __restrict__ outF, __bf16* __restrict__ outB,
    int M, int N, int K) {
  __shared__ __bf16 sA[2][128 * LDST];
  __shared__ __bf16 sB[2][128 * LDST];
  const int tid = threadIdx.x, lane = tid & 31, wave = tid >> 5;
  const int wm = wave >> 2, wn = wave & 3;
  const int mbase = blockIdx.y * 128, nbase = blockIdx.x * 128;
  const int l16 = lane & 15;
  const int kloA = (lane & 16) ? 8 : 0;    // A-operand K offset per lane half
  const int kloB = (lane & 16) ? 8 : 0;    // B-operand K offset per lane half
  v8f acc[4][2] = {};

  const __bf16* Atile = A + (size_t)mbase * K;
  const __bf16* Btile = Bt + (size_t)nbase * K;
  const int nK = K >> 5;

  if (wave == 0) {
    tdm_load_tile((unsigned)(size_t)(void*)&sA[0][0], Atile, K, M);
    tdm_load_tile((unsigned)(size_t)(void*)&sB[0][0], Btile, K, N);
  }

  for (int kb = 0; kb < nK; ++kb) {
    const int buf = kb & 1;
    if (wave == 0) {
      if (kb + 1 < nK) {
        tdm_load_tile((unsigned)(size_t)(void*)&sA[buf ^ 1][0],
                      Atile + (kb + 1) * 32, K, M);
        tdm_load_tile((unsigned)(size_t)(void*)&sB[buf ^ 1][0],
                      Btile + (kb + 1) * 32, K, N);
        __builtin_amdgcn_s_wait_tensorcnt(2);   // current tile's 2 DMAs done
      } else {
        __builtin_amdgcn_s_wait_tensorcnt(0);
      }
    }
    __syncthreads();

    v16bf af[4], bfv[2];
#pragma unroll
    for (int im = 0; im < 4; ++im) {
      const int r = wm * 64 + im * 16 + l16;
      U16 u;
      u.h[0] = *(const v8bf*)&sA[buf][r * LDST + kloA];
      u.h[1] = *(const v8bf*)&sA[buf][r * LDST + 16 + kloA];
      af[im] = u.v;
    }
#pragma unroll
    for (int in = 0; in < 2; ++in) {
      const int rc = wn * 32 + in * 16 + l16;
      U16 u;
      u.h[0] = *(const v8bf*)&sB[buf][rc * LDST + kloB];
      u.h[1] = *(const v8bf*)&sB[buf][rc * LDST + 16 + kloB];
      bfv[in] = u.v;
    }
#pragma unroll
    for (int im = 0; im < 4; ++im)
#pragma unroll
      for (int in = 0; in < 2; ++in)
        acc[im][in] = __builtin_amdgcn_wmma_f32_16x16x32_bf16(
            false, af[im], false, bfv[in], (short)0, acc[im][in], false, false);
    __syncthreads();   // all waves done reading buf before TDM overwrites it
  }

  const int rsel = (lane & 16) ? 8 : 0;
#pragma unroll
  for (int im = 0; im < 4; ++im)
#pragma unroll
    for (int in = 0; in < 2; ++in) {
      const int col = nbase + wn * 32 + in * 16 + l16;
#pragma unroll
      for (int r = 0; r < 8; ++r) {
        const int row = mbase + wm * 64 + im * 16 + rsel + r;
        const size_t idx = (size_t)row * N + col;
        const float v = acc[im][in][r];
        if constexpr (EPI == 0) {
          outB[idx] = (__bf16)v;
        } else if constexpr (EPI == 1) {
          outF[idx] = v + bias[col] + resid[idx];
        } else {
          const float t = v + bias[col];
          outB[idx] = (__bf16)(t / (1.0f + __expf(-t)));   // silu
        }
      }
    }
}

// ---------------------------------------------------------------------------
// Flash attention: qkv bf16 [TOKENS, 3*DIM] -> o bf16 [TOKENS, DIM]
// grid.x = SEQ/128, grid.y = BATCH*NHEADS; each wave = 16 query rows
// ---------------------------------------------------------------------------
__global__ __launch_bounds__(256) void flash_attn_wmma(
    const __bf16* __restrict__ qkv, __bf16* __restrict__ obuf) {
  __shared__ __bf16 pbuf[8 * 16 * 32];
  const int tid = threadIdx.x, lane = tid & 31, wave = tid >> 5;
  const int l16 = lane & 15;
  const int b = blockIdx.y >> 4, h = blockIdx.y & 15;
  const int q0 = blockIdx.x * 128 + wave * 16;
  const int kloA = (lane & 16) ? 8 : 0;
  const int kbB  = (lane & 16) ? 8 : 0;
  __bf16* pb = &pbuf[wave * 512];

  // Q fragments (A-operand), 2 chunks of K=32 over Dh=64
  v16bf qf[2];
  {
    const __bf16* qptr = qkv + (size_t)(b * SEQ + q0 + l16) * (3 * DIM) + h * HDIM;
#pragma unroll
    for (int kc = 0; kc < 2; ++kc) {
      U16 u;
      u.h[0] = *(const v8bf*)(qptr + kc * 32 + kloA);
      u.h[1] = *(const v8bf*)(qptr + kc * 32 + 16 + kloA);
      qf[kc] = u.v;
    }
  }

  v8f oacc[4] = {};
  float mrow[8], lrow[8];
#pragma unroll
  for (int r = 0; r < 8; ++r) { mrow[r] = -1e30f; lrow[r] = 0.f; }
  const float scale = 0.125f;   // 1/sqrt(64)
  const int prow = (lane & 16) ? 8 : 0;

  for (int kt = 0; kt < SEQ; kt += 32) {
    // scores 16x32 = two 16x16 accumulators
    v8f sc[2] = {};
#pragma unroll
    for (int nt = 0; nt < 2; ++nt) {
      const __bf16* kptr =
          qkv + (size_t)(b * SEQ + kt + nt * 16 + l16) * (3 * DIM) + DIM + h * HDIM;
#pragma unroll
      for (int kc = 0; kc < 2; ++kc) {
        U16 u;
        u.h[0] = *(const v8bf*)(kptr + kc * 32 + kbB);
        u.h[1] = *(const v8bf*)(kptr + kc * 32 + kbB + 16);
        sc[nt] = __builtin_amdgcn_wmma_f32_16x16x32_bf16(
            false, qf[kc], false, u.v, (short)0, sc[nt], false, false);
      }
    }
    // online softmax (row reductions within 16-lane halves)
    float alpha[8];
#pragma unroll
    for (int r = 0; r < 8; ++r) {
      float t = fmaxf(sc[0][r], sc[1][r]) * scale;
      t = fmaxf(t, __shfl_xor(t, 1, 16));
      t = fmaxf(t, __shfl_xor(t, 2, 16));
      t = fmaxf(t, __shfl_xor(t, 4, 16));
      t = fmaxf(t, __shfl_xor(t, 8, 16));
      const float mnew = fmaxf(mrow[r], t);
      alpha[r] = __expf(mrow[r] - mnew);
      mrow[r] = mnew;
    }
#pragma unroll
    for (int r = 0; r < 8; ++r) {
      const float p0 = __expf(sc[0][r] * scale - mrow[r]);
      const float p1 = __expf(sc[1][r] * scale - mrow[r]);
      sc[0][r] = p0; sc[1][r] = p1;
      float s = p0 + p1;
      s += __shfl_xor(s, 1, 16);
      s += __shfl_xor(s, 2, 16);
      s += __shfl_xor(s, 4, 16);
      s += __shfl_xor(s, 8, 16);
      lrow[r] = lrow[r] * alpha[r] + s;
    }
#pragma unroll
    for (int t4 = 0; t4 < 4; ++t4)
#pragma unroll
      for (int r = 0; r < 8; ++r) oacc[t4][r] *= alpha[r];

    // P: C-layout -> A-layout via per-wave LDS staging
#pragma unroll
    for (int nt = 0; nt < 2; ++nt)
#pragma unroll
      for (int r = 0; r < 8; ++r)
        pb[(prow + r) * 32 + nt * 16 + l16] = (__bf16)sc[nt][r];
    asm volatile("s_wait_dscnt 0" ::: "memory");
    v16bf pf;
    {
      U16 u;
      u.h[0] = *(const v8bf*)&pb[l16 * 32 + kloA];
      u.h[1] = *(const v8bf*)&pb[l16 * 32 + 16 + kloA];
      pf = u.v;
    }
    // P @ V : V as B-operand (gather per lane, coalesced across lanes)
#pragma unroll
    for (int t4 = 0; t4 < 4; ++t4) {
      const __bf16* vptr = qkv + (size_t)(b * SEQ + kt + kbB) * (3 * DIM) +
                           2 * DIM + h * HDIM + t4 * 16 + l16;
      v16bf vf;
#pragma unroll
      for (int j = 0; j < 16; ++j) vf[j] = vptr[(size_t)(j + (j & 8)) * (3 * DIM)];
      oacc[t4] = __builtin_amdgcn_wmma_f32_16x16x32_bf16(
          false, pf, false, vf, (short)0, oacc[t4], false, false);
    }
  }
  // write o (token-major, heads concatenated)
#pragma unroll
  for (int t4 = 0; t4 < 4; ++t4)
#pragma unroll
    for (int r = 0; r < 8; ++r) {
      const int srow = q0 + prow + r;
      obuf[(size_t)(b * SEQ + srow) * DIM + h * HDIM + t4 * 16 + l16] =
          (__bf16)(oacc[t4][r] / lrow[r]);
    }
}

// ---------------------------------------------------------------------------
extern "C" void kernel_launch(void* const* d_in, const int* in_sizes, int n_in,
                              void* d_out, int out_size, void* d_ws, size_t ws_size,
                              hipStream_t stream) {
  (void)in_sizes; (void)n_in; (void)out_size; (void)ws_size;
  const float* x       = (const float*)d_in[0];
  const float* norm1_w = (const float*)d_in[1];
  const float* qkv_w   = (const float*)d_in[2];
  const float* proj_w  = (const float*)d_in[3];
  const float* proj_b  = (const float*)d_in[4];
  const float* norm2_w = (const float*)d_in[5];
  const float* fc1_w   = (const float*)d_in[6];
  const float* fc1_b   = (const float*)d_in[7];
  const float* fc2_w   = (const float*)d_in[8];
  const float* fc2_b   = (const float*)d_in[9];
  float* out = (float*)d_out;

  char* ws = (char*)d_ws;
  size_t off = 0;
  auto alloc = [&](size_t bytes) -> void* {
    void* p = ws + off;
    off += (bytes + 255) & ~(size_t)255;
    return p;
  };
  __bf16* qkvw_t  = (__bf16*)alloc((size_t)3 * DIM * DIM * 2);       // [3072,1024]
  __bf16* projw_t = (__bf16*)alloc((size_t)DIM * DIM * 2);           // [1024,1024]
  __bf16* fc1w_t  = (__bf16*)alloc((size_t)MLPDIM * DIM * 2);        // [4096,1024]
  __bf16* fc2w_t  = (__bf16*)alloc((size_t)DIM * MLPDIM * 2);        // [1024,4096]
  __bf16* h       = (__bf16*)alloc((size_t)TOKENS * DIM * 2);
  __bf16* qkvbuf  = (__bf16*)alloc((size_t)TOKENS * 3 * DIM * 2);
  __bf16* obuf    = (__bf16*)alloc((size_t)TOKENS * DIM * 2);
  float*  x1      = (float*) alloc((size_t)TOKENS * DIM * 4);
  __bf16* h2      = (__bf16*)alloc((size_t)TOKENS * DIM * 2);
  __bf16* m1      = (__bf16*)alloc((size_t)TOKENS * MLPDIM * 2);

  // 1) weights -> bf16 transposed [N,K]
  transpose_w_bf16<<<dim3(3 * DIM / 32, DIM / 32), 256, 0, stream>>>(qkv_w, qkvw_t, DIM, 3 * DIM);
  transpose_w_bf16<<<dim3(DIM / 32, DIM / 32), 256, 0, stream>>>(proj_w, projw_t, DIM, DIM);
  transpose_w_bf16<<<dim3(MLPDIM / 32, DIM / 32), 256, 0, stream>>>(fc1_w, fc1w_t, DIM, MLPDIM);
  transpose_w_bf16<<<dim3(DIM / 32, MLPDIM / 32), 256, 0, stream>>>(fc2_w, fc2w_t, MLPDIM, DIM);

  // 2) h = rmsnorm(x)
  rmsnorm_bf16<<<TOKENS, 256, 0, stream>>>(x, norm1_w, h);

  // 3) qkv = h @ qkv_w   (no bias)
  gemm_bf16_wmma<0><<<dim3(3 * DIM / 128, TOKENS / 128), 256, 0, stream>>>(
      h, qkvw_t, nullptr, nullptr, nullptr, qkvbuf, TOKENS, 3 * DIM, DIM);

  // 4) attention
  flash_attn_wmma<<<dim3(SEQ / 128, BATCH * NHEADS), 256, 0, stream>>>(qkvbuf, obuf);

  // 5) x1 = x + o @ proj_w + proj_b
  gemm_bf16_wmma<1><<<dim3(DIM / 128, TOKENS / 128), 256, 0, stream>>>(
      obuf, projw_t, proj_b, x, x1, nullptr, TOKENS, DIM, DIM);

  // 6) h2 = rmsnorm(x1)
  rmsnorm_bf16<<<TOKENS, 256, 0, stream>>>(x1, norm2_w, h2);

  // 7) m1 = silu(h2 @ fc1_w + fc1_b)
  gemm_bf16_wmma<2><<<dim3(MLPDIM / 128, TOKENS / 128), 256, 0, stream>>>(
      h2, fc1w_t, fc1_b, nullptr, nullptr, m1, TOKENS, MLPDIM, DIM);

  // 8) out = x1 + m1 @ fc2_w + fc2_b
  gemm_bf16_wmma<1><<<dim3(DIM / 128, TOKENS / 128), 256, 0, stream>>>(
      m1, fc2w_t, fc2_b, x1, out, nullptr, TOKENS, DIM, MLPDIM);
}
